// VectorQuantizer_11347303596244
// MI455X (gfx1250) — compile-verified, matrix operand order checked by reading, not run on hardware
//
#include <hip/hip_runtime.h>
#include <hip/hip_bf16.h>

#define N_Z    65536
#define K_W    8192
#define D_DIM  256
#define TILE_M 128
#define CHUNK_K 64
#define LDS_STRIDE 264   // 256 bf16 + 8 pad (keeps 16B alignment, spreads banks)

typedef __bf16 v16bf __attribute__((ext_vector_type(16)));
typedef float  v8f   __attribute__((ext_vector_type(8)));

union FragU { v16bf v; uint4 q[2]; };
union PackU { __bf16 h[4]; uint2 u2; };

// ---------------------------------------------------------------------------
// Kernel 1: fused (z @ W^T) via v_wmma_f32_16x16x32_bf16 with argmin epilogue.
// dist = ||W_k||^2 - 2 * z.W_k   (||z||^2 is constant per row -> irrelevant).
// ---------------------------------------------------------------------------
__global__ __launch_bounds__(256) void vq_argmin_kernel(
    const float* __restrict__ z, const float* __restrict__ W,
    int* __restrict__ idx_out, float* __restrict__ idx_out_f)
{
    __shared__ __bf16 lds_w[CHUNK_K][LDS_STRIDE];
    __shared__ float  w2[CHUNK_K];

    const int tid  = threadIdx.x;
    const int wave = tid >> 5;
    const int lane = tid & 31;
    const int lrow = lane & 15;   // row-in-16 for A/B frags, N column for C
    const int half = lane >> 4;   // K-half selector for A/B frags

    // ---- A fragments: this wave's 16 z rows, bf16, kept in VGPRs ----
    const int m0 = blockIdx.x * TILE_M + wave * 16;
    const float* zr = z + (size_t)(m0 + lrow) * D_DIM;
    FragU afrag[8];
#pragma unroll
    for (int f = 0; f < 8; ++f) {
        const int k0 = f * 32 + half * 8;
        float4 a0 = *reinterpret_cast<const float4*>(zr + k0);
        float4 a1 = *reinterpret_cast<const float4*>(zr + k0 + 4);
        float4 a2 = *reinterpret_cast<const float4*>(zr + k0 + 16);
        float4 a3 = *reinterpret_cast<const float4*>(zr + k0 + 20);
        afrag[f].v[0]  = (__bf16)a0.x; afrag[f].v[1]  = (__bf16)a0.y;
        afrag[f].v[2]  = (__bf16)a0.z; afrag[f].v[3]  = (__bf16)a0.w;
        afrag[f].v[4]  = (__bf16)a1.x; afrag[f].v[5]  = (__bf16)a1.y;
        afrag[f].v[6]  = (__bf16)a1.z; afrag[f].v[7]  = (__bf16)a1.w;
        afrag[f].v[8]  = (__bf16)a2.x; afrag[f].v[9]  = (__bf16)a2.y;
        afrag[f].v[10] = (__bf16)a2.z; afrag[f].v[11] = (__bf16)a2.w;
        afrag[f].v[12] = (__bf16)a3.x; afrag[f].v[13] = (__bf16)a3.y;
        afrag[f].v[14] = (__bf16)a3.z; afrag[f].v[15] = (__bf16)a3.w;
    }

    float rmin[8];
    unsigned int ridx[8];
#pragma unroll
    for (int v = 0; v < 8; ++v) { rmin[v] = 3.4e38f; ridx[v] = 0u; }

    for (int kc = 0; kc < K_W; kc += CHUNK_K) {
        // ---- cooperative load: 64 W rows -> bf16 LDS, plus ||W_k||^2 ----
        {
            const int r = tid >> 2;          // chunk row 0..63
            const int p = tid & 3;           // column quarter
            const float* wr = W + (size_t)(kc + r) * D_DIM + p * 64;
            float sq = 0.f;
#pragma unroll
            for (int i = 0; i < 16; ++i) {
                float4 x = *reinterpret_cast<const float4*>(wr + i * 4);
                sq += x.x * x.x + x.y * x.y + x.z * x.z + x.w * x.w;
                PackU pk;
                pk.h[0] = (__bf16)x.x; pk.h[1] = (__bf16)x.y;
                pk.h[2] = (__bf16)x.z; pk.h[3] = (__bf16)x.w;
                *reinterpret_cast<uint2*>(&lds_w[r][p * 64 + i * 4]) = pk.u2;
            }
            sq += __shfl_xor(sq, 1, 32);
            sq += __shfl_xor(sq, 2, 32);
            if (p == 0) w2[r] = sq;
            if (kc + CHUNK_K < K_W)
                __builtin_prefetch(wr + (size_t)CHUNK_K * D_DIM, 0, 3);
        }
        __syncthreads();

        float w2l[4];
#pragma unroll
        for (int j = 0; j < 4; ++j) w2l[j] = w2[j * 16 + lrow];

        v8f acc[4];
#pragma unroll
        for (int j = 0; j < 4; ++j)
#pragma unroll
            for (int e = 0; e < 8; ++e) acc[j][e] = 0.f;

        // ---- 8 K-steps x 4 N-tiles of v_wmma_f32_16x16x32_bf16 ----
#pragma unroll
        for (int f = 0; f < 8; ++f) {
            const int k0 = f * 32 + half * 8;
#pragma unroll
            for (int j = 0; j < 4; ++j) {
                FragU b;
                const __bf16* bp = &lds_w[j * 16 + lrow][k0];
                b.q[0] = *reinterpret_cast<const uint4*>(bp);
                b.q[1] = *reinterpret_cast<const uint4*>(bp + 16);
                acc[j] = __builtin_amdgcn_wmma_f32_16x16x32_bf16(
                    false, afrag[f].v, false, b.v, (short)0, acc[j],
                    false, false);
            }
        }

        // ---- epilogue: running (min, argmin) from accumulator layout ----
        // C layout: lane = N column (lrow), VGPR v -> M = v + 8*half.
#pragma unroll
        for (int j = 0; j < 4; ++j) {
            const unsigned int kidx = (unsigned int)(kc + j * 16 + lrow);
#pragma unroll
            for (int v = 0; v < 8; ++v) {
                float dist = w2l[j] - 2.0f * acc[j][v];
                if (dist < rmin[v]) { rmin[v] = dist; ridx[v] = kidx; }
            }
        }
        __syncthreads();
    }

    // ---- cross-lane reduction within each 16-lane half (tie -> lowest k) ----
#pragma unroll
    for (int v = 0; v < 8; ++v) {
        float m = rmin[v];
        unsigned int ix = ridx[v];
#pragma unroll
        for (int s = 1; s < 16; s <<= 1) {
            float om        = __shfl_xor(m, s, 32);
            unsigned int oi = __shfl_xor(ix, s, 32);
            if (om < m || (om == m && oi < ix)) { m = om; ix = oi; }
        }
        if (lrow == 0) {
            const int row = m0 + half * 8 + v;
            idx_out[row]   = (int)ix;
            idx_out_f[row] = (float)ix;
        }
    }
}

// ---------------------------------------------------------------------------
// Kernel 2: quantized_st = W[idx] (exact fp32 gather) + loss accumulation.
// loss = 1.25 * mean((W[idx] - z)^2)
// ---------------------------------------------------------------------------
__global__ __launch_bounds__(256) void vq_gather_loss_kernel(
    const float* __restrict__ z, const float* __restrict__ W,
    const int* __restrict__ idx, float* __restrict__ quant,
    float* __restrict__ loss)
{
    const int row = blockIdx.x;
    const int d   = threadIdx.x;
    const int k   = idx[row];
    const float w  = W[(size_t)k * D_DIM + d];
    const float zv = z[(size_t)row * D_DIM + d];
    quant[(size_t)row * D_DIM + d] = w;
    const float diff = w - zv;
    float val = diff * diff;
#pragma unroll
    for (int s = 16; s >= 1; s >>= 1) val += __shfl_xor(val, s, 32);

    __shared__ float partial[8];
    const int wave = threadIdx.x >> 5;
    const int lane = threadIdx.x & 31;
    if (lane == 0) partial[wave] = val;
    __syncthreads();
    if (threadIdx.x == 0) {
        float s = 0.f;
#pragma unroll
        for (int i = 0; i < 8; ++i) s += partial[i];
        unsafeAtomicAdd(loss, s * (1.25f / ((float)N_Z * (float)D_DIM)));
    }
}

__global__ void vq_init_kernel(float* __restrict__ loss) { *loss = 0.f; }

// ---------------------------------------------------------------------------
extern "C" void kernel_launch(void* const* d_in, const int* in_sizes, int n_in,
                              void* d_out, int out_size, void* d_ws, size_t ws_size,
                              hipStream_t stream) {
    const float* z = (const float*)d_in[0];   // [65536, 256] fp32
    const float* W = (const float*)d_in[1];   // [8192, 256] fp32

    float* out   = (float*)d_out;
    float* quant = out;                                  // [65536*256]
    float* loss  = out + (size_t)N_Z * D_DIM;            // [1]
    float* idx_f = loss + 1;                             // [65536] as float
    int*   idx_i = (int*)d_ws;                           // scratch int indices

    vq_init_kernel<<<1, 1, 0, stream>>>(loss);
    vq_argmin_kernel<<<N_Z / TILE_M, 256, 0, stream>>>(z, W, idx_i, idx_f);
    vq_gather_loss_kernel<<<N_Z, 256, 0, stream>>>(z, W, idx_i, quant, loss);
}
